// DINOBevAligner_40879498729129
// MI455X (gfx1250) — matrix operand order repaired and checked by reading, not compiled, not run on hardware
//
#include <hip/hip_runtime.h>
#include <hip/hip_bf16.h>
#include <math.h>

// ---------------- types ----------------
typedef __attribute__((ext_vector_type(16))) _Float16 v16h;
typedef __attribute__((ext_vector_type(8)))  _Float16 v8h;
typedef __attribute__((ext_vector_type(4)))  _Float16 v4h;
typedef __attribute__((ext_vector_type(8)))  float    v8f;

// ---------------- problem constants ----------------
#define NV    6
#define HP    37
#define WP    37
#define NC    768
#define QN    6400      // 80*80
#define ND    4
#define CCTX  256
#define NG    3

// geometry constants
#define SCALE_F   0.32375f
#define PAD_T_F   113.0f
#define PAD_L_F   0.0f
#define W2M1      517.0f   // W2-1
#define H2M1      517.0f   // H2-1

__device__ __forceinline__ float softplus_f(float x) {
    // jax.nn.softplus = log1p(exp(x)); stable for large x
    return (x > 20.0f) ? x : log1pf(expf(x));
}

// =====================================================================
// Kernel 1: project + bilinear sample + LayerNorm + depth/view fusion.
// One wave32 per BEV query. Lane covers channels c = chunk*128 + lane*4 .. +3
// (6 float4 chunks = 768 channels). Output: f_bev as f16 (QN x NC) row-major.
// =====================================================================
__global__ __launch_bounds__(256) void k_sample_fuse(
    const float* __restrict__ tok,    // (V, HP*WP, NC) f32
    const float* __restrict__ l2i,    // (V, 4, 4) f32
    const float* __restrict__ wview,  // (V) f32
    const float* __restrict__ gamma,  // (NC)
    const float* __restrict__ beta,   // (NC)
    _Float16*   __restrict__ outA)    // (QN, NC) f16
{
    const int lane = threadIdx.x & 31;
    const int wid  = (blockIdx.x * blockDim.x + threadIdx.x) >> 5;
    if (wid >= QN) return;
    const int q  = wid;
    const int qx = q % 80;
    const int qy = q / 80;
    const float X = (qx + 0.5f) * 1.28f - 51.2f;   // xs*(102.4) - 51.2
    const float Y = (qy + 0.5f) * 1.28f - 51.2f;

    float4 num[6];
#pragma unroll
    for (int i = 0; i < 6; ++i) num[i] = make_float4(0.f, 0.f, 0.f, 0.f);
    float den = 0.f;

    for (int v = 0; v < NV; ++v) {
        const float* L = l2i + v * 16;
        const float L00 = L[0],  L01 = L[1],  L02 = L[2],  L03 = L[3];
        const float L10 = L[4],  L11 = L[5],  L12 = L[6],  L13 = L[7];
        const float L20 = L[8],  L21 = L[9],  L22 = L[10], L23 = L[11];
        const float* fbase = tok + (size_t)v * (HP * WP * NC);

        float4 accd[6];
#pragma unroll
        for (int i = 0; i < 6; ++i) accd[i] = make_float4(0.f, 0.f, 0.f, 0.f);
        float cnt = 0.f;

        for (int d = 0; d < ND; ++d) {
            const float Zc = (float)d * (7.0f / 3.0f) - 4.5f;  // linspace(0.5,7.5,4)-5
            const float c0 = L00 * X + L01 * Y + L02 * Zc + L03;
            const float c1 = L10 * X + L11 * Y + L12 * Zc + L13;
            const float c2 = L20 * X + L21 * Y + L22 * Zc + L23;
            const bool  dm = (c2 > 1e-5f);
            const float invd = 1.0f / fmaxf(c2, 1e-5f);
            const float ud = c0 * invd * SCALE_F + PAD_L_F;
            const float vd = c1 * invd * SCALE_F + PAD_T_F;
            const bool valid = dm && (ud >= 0.f) && (ud <= W2M1) &&
                                     (vd >= 0.f) && (vd <= H2M1);
            if (!valid) continue;   // wave-uniform branch
            cnt += 1.0f;

            const float x = ud * (36.0f / 517.0f);
            const float y = vd * (36.0f / 517.0f);
            const float x0f = floorf(x), y0f = floorf(y);
            const float wx = x - x0f, wy = y - y0f;
            const int xi = (int)x0f, yi = (int)y0f;

            // 4 corners with per-corner bounds check (zero padding)
            const bool ib00 = (xi   >= 0) && (xi   <= WP - 1) && (yi   >= 0) && (yi   <= HP - 1);
            const bool ib10 = (xi+1 >= 0) && (xi+1 <= WP - 1) && (yi   >= 0) && (yi   <= HP - 1);
            const bool ib01 = (xi   >= 0) && (xi   <= WP - 1) && (yi+1 >= 0) && (yi+1 <= HP - 1);
            const bool ib11 = (xi+1 >= 0) && (xi+1 <= WP - 1) && (yi+1 >= 0) && (yi+1 <= HP - 1);
            const int cx0 = min(max(xi,     0), WP - 1);
            const int cx1 = min(max(xi + 1, 0), WP - 1);
            const int cy0 = min(max(yi,     0), HP - 1);
            const int cy1 = min(max(yi + 1, 0), HP - 1);
            const float w0 = ib00 ? (1.f - wx) * (1.f - wy) : 0.f;
            const float w1 = ib10 ? wx * (1.f - wy)         : 0.f;
            const float w2 = ib01 ? (1.f - wx) * wy         : 0.f;
            const float w3 = ib11 ? wx * wy                 : 0.f;
            const float* p0 = fbase + (size_t)(cy0 * WP + cx0) * NC;
            const float* p1 = fbase + (size_t)(cy0 * WP + cx1) * NC;
            const float* p2 = fbase + (size_t)(cy1 * WP + cx0) * NC;
            const float* p3 = fbase + (size_t)(cy1 * WP + cx1) * NC;

            float4 pt[6];
            float lsum = 0.f, lsq = 0.f;
#pragma unroll
            for (int ch = 0; ch < 6; ++ch) {
                const int co = ch * 128 + lane * 4;
                const float4 f0 = *(const float4*)(p0 + co);
                const float4 f1 = *(const float4*)(p1 + co);
                const float4 f2 = *(const float4*)(p2 + co);
                const float4 f3 = *(const float4*)(p3 + co);
                float4 s;
                s.x = w0 * f0.x + w1 * f1.x + w2 * f2.x + w3 * f3.x;
                s.y = w0 * f0.y + w1 * f1.y + w2 * f2.y + w3 * f3.y;
                s.z = w0 * f0.z + w1 * f1.z + w2 * f2.z + w3 * f3.z;
                s.w = w0 * f0.w + w1 * f1.w + w2 * f2.w + w3 * f3.w;
                pt[ch] = s;
                lsum += s.x + s.y + s.z + s.w;
                lsq  += s.x * s.x + s.y * s.y + s.z * s.z + s.w * s.w;
            }
            // wave32 reduction for LayerNorm statistics
#pragma unroll
            for (int off = 16; off > 0; off >>= 1) {
                lsum += __shfl_xor(lsum, off, 32);
                lsq  += __shfl_xor(lsq,  off, 32);
            }
            const float mu   = lsum * (1.0f / NC);
            const float var  = lsq * (1.0f / NC) - mu * mu;   // population var
            const float rstd = rsqrtf(var + 1e-5f);
#pragma unroll
            for (int ch = 0; ch < 6; ++ch) {
                const int co = ch * 128 + lane * 4;
                const float4 g4 = *(const float4*)(gamma + co);
                const float4 b4 = *(const float4*)(beta + co);
                accd[ch].x += (pt[ch].x - mu) * rstd * g4.x + b4.x;
                accd[ch].y += (pt[ch].y - mu) * rstd * g4.y + b4.y;
                accd[ch].z += (pt[ch].z - mu) * rstd * g4.z + b4.z;
                accd[ch].w += (pt[ch].w - mu) * rstd * g4.w + b4.w;
            }
        }

        const float denom = fmaxf(cnt, 1e-6f);
        const float wv    = softplus_f(wview[v]);
        const float s     = wv / denom;
#pragma unroll
        for (int ch = 0; ch < 6; ++ch) {
            num[ch].x += s * accd[ch].x;
            num[ch].y += s * accd[ch].y;
            num[ch].z += s * accd[ch].z;
            num[ch].w += s * accd[ch].w;
        }
        // reference: view_valid = (max(cnt,EPS) > 0) == always 1
        den += wv;
    }

    const float invden = 1.0f / fmaxf(den, 1e-6f);
#pragma unroll
    for (int ch = 0; ch < 6; ++ch) {
        const int co = ch * 128 + lane * 4;
        v4h h;
        h[0] = (_Float16)(num[ch].x * invden);
        h[1] = (_Float16)(num[ch].y * invden);
        h[2] = (_Float16)(num[ch].z * invden);
        h[3] = (_Float16)(num[ch].w * invden);
        *(v4h*)(outA + (size_t)q * NC + co) = h;
    }
}

// =====================================================================
// Kernel 2: build dense block-diagonal group-softmax weight matrix
// Wg (CCTX x NC) f16, Wg[j, c] = softmax(logits[j,:])[c-3j] if c in group j.
// =====================================================================
__global__ void k_build_wg(const float* __restrict__ logits,  // (CCTX, NG)
                           _Float16*   __restrict__ Wg)       // (CCTX, NC)
{
    const int idx = blockIdx.x * blockDim.x + threadIdx.x;
    if (idx >= CCTX * NC) return;
    const int j = idx / NC;
    const int c = idx - j * NC;
    const int g = c - 3 * j;
    float val = 0.f;
    if (g >= 0 && g < NG) {
        const float l0 = logits[j * NG + 0];
        const float l1 = logits[j * NG + 1];
        const float l2 = logits[j * NG + 2];
        const float mx = fmaxf(l0, fmaxf(l1, l2));
        const float e0 = expf(l0 - mx), e1 = expf(l1 - mx), e2 = expf(l2 - mx);
        const float inv = 1.0f / (e0 + e1 + e2);
        const float eg = (g == 0) ? e0 : ((g == 1) ? e1 : e2);
        val = eg * inv;
    }
    Wg[idx] = (_Float16)val;
}

// =====================================================================
// Kernel 3: D(256 x 6400) = Wg(256 x 768) * f_bev^T(768 x 6400) via
// v_wmma_f32_16x16x32_f16, one wave per 16x16 output tile, K = 768.
// A fragment (M=j, K): lane (half=l>>4, m=l&15), elem i -> K = 16*half + i (contiguous)
// B fragment (K, N=q): lane (half, n=l&15), elem i -> K = 16*half + i (contiguous)
// D fragment: VGPR r -> row m = 8*half + r, col n = l&15  (n maps to contiguous q)
// =====================================================================
__global__ __launch_bounds__(256) void k_group_gemm(
    const _Float16* __restrict__ Wg,   // (CCTX, NC)
    const _Float16* __restrict__ Fb,   // (QN, NC)
    float*          __restrict__ out)  // (CCTX, QN)  == (B, C_CTX, 80, 80)
{
    const int lane = threadIdx.x & 31;
    const int wid  = (blockIdx.x * blockDim.x + threadIdx.x) >> 5;
    const int NTQ  = QN / 16;            // 400
    if (wid >= (CCTX / 16) * NTQ) return;
    const int jt = wid / NTQ;
    const int qt = wid - jt * NTQ;
    const int j0 = jt * 16;
    const int q0 = qt * 16;

    const int half = lane >> 4;
    const int mn   = lane & 15;

    const _Float16* arow = Wg + (size_t)(j0 + mn) * NC + 16 * half;
    const _Float16* brow = Fb + (size_t)(q0 + mn) * NC + 16 * half;

    v8f acc = {};
#pragma unroll 4
    for (int k0 = 0; k0 < NC; k0 += 32) {
        const v16h a = *(const v16h*)(arow + k0);        // K = k0 + 16*half + 0..15
        const v16h b = *(const v16h*)(brow + k0);        // K = k0 + 16*half + 0..15
        acc = __builtin_amdgcn_wmma_f32_16x16x32_f16(
                  /*neg_a=*/false, a, /*neg_b=*/false, b,
                  /*c_mod=*/(short)0, acc, /*reuse_a=*/false, /*reuse_b=*/false);
    }

#pragma unroll
    for (int r = 0; r < 8; ++r) {
        const int m = half * 8 + r;
        out[(size_t)(j0 + m) * QN + q0 + mn] = acc[r];
    }
}

// =====================================================================
// Launch
// =====================================================================
extern "C" void kernel_launch(void* const* d_in, const int* in_sizes, int n_in,
                              void* d_out, int out_size, void* d_ws, size_t ws_size,
                              hipStream_t stream) {
    const float* tok    = (const float*)d_in[0];  // last_tokens (1,6,1369,768)
    const float* l2i    = (const float*)d_in[1];  // lidar2img   (1,6,4,4)
    const float* wv     = (const float*)d_in[2];  // w_view      (1,6,1)
    const float* gam    = (const float*)d_in[3];  // ln_gamma    (768)
    const float* bet    = (const float*)d_in[4];  // ln_beta     (768)
    const float* logits = (const float*)d_in[5];  // logits      (256,3)
    float*       out    = (float*)d_out;          // (1,256,80,80)

    char* ws = (char*)d_ws;
    _Float16* Fb = (_Float16*)ws;                                        // 9,830,400 B
    _Float16* Wg = (_Float16*)(ws + (size_t)QN * NC * sizeof(_Float16)); //   393,216 B

    // K1: 6400 waves (one per query), 8 waves/block
    k_sample_fuse<<<QN / 8, 256, 0, stream>>>(tok, l2i, wv, gam, bet, Fb);
    // K2: build dense group-softmax weight matrix
    k_build_wg<<<(CCTX * NC + 255) / 256, 256, 0, stream>>>(logits, Wg);
    // K3: 16x400 = 6400 tiles, 8 waves/block
    k_group_gemm<<<((CCTX / 16) * (QN / 16)) / 8, 256, 0, stream>>>(Wg, Fb, out);
}
